// EncoderLayer_11854109737419
// MI455X (gfx1250) — matrix operand order repaired and checked by reading, not compiled, not run on hardware
//
#include <hip/hip_runtime.h>
#include <hip/hip_bf16.h>

// ---------------------------------------------------------------------------
// CDNA5 / gfx1250 encoder layer: all GEMMs via v_wmma_f32_16x16x32_bf16
// (bf16 inputs, fp32 accumulate), A-tile staging via async global->LDS DMA
// (GLOBAL_LOAD_ASYNC_TO_LDS_B128, ASYNCcnt path). wave32 throughout.
// ---------------------------------------------------------------------------

typedef __attribute__((ext_vector_type(16))) __bf16 v16bf;
typedef __attribute__((ext_vector_type(8)))  float  v8f;

__device__ __forceinline__ unsigned short f2bf(float f) {
  // round-to-nearest-even fp32 -> bf16
  unsigned u = __float_as_uint(f);
  u += 0x7FFFu + ((u >> 16) & 1u);
  return (unsigned short)(u >> 16);
}
__device__ __forceinline__ float bf2f(unsigned short h) {
  return __uint_as_float(((unsigned)h) << 16);
}
__device__ __forceinline__ __bf16 bfbits(unsigned short h) {
  return __builtin_bit_cast(__bf16, h);
}

// ---- async global -> LDS (gfx1250 GLOBAL_LOAD_ASYNC_TO_LDS_B128, ASYNCcnt)
#if __has_builtin(__builtin_amdgcn_global_load_async_to_lds_b128)
#define HAVE_ASYNC_LDS 1
typedef int v4i_vs __attribute__((vector_size(16)));
typedef __attribute__((address_space(1))) v4i_vs* gp_v4i;
typedef __attribute__((address_space(3))) v4i_vs* lp_v4i;

__device__ __forceinline__ void async_copy_b128(void* lds, const void* g) {
  __builtin_amdgcn_global_load_async_to_lds_b128((gp_v4i)g, (lp_v4i)lds, 0, 0);
}
__device__ __forceinline__ void wait_async0() {
#if __has_builtin(__builtin_amdgcn_s_wait_asynccnt)
  __builtin_amdgcn_s_wait_asynccnt(0);
#else
  asm volatile("s_wait_asynccnt 0" ::: "memory");
#endif
}
#else
#define HAVE_ASYNC_LDS 0
#endif

// ===========================================================================
// Generic GEMM: C[M,N] = act(A[M,K] @ W[K,N] + bias[N]),  A,W,bias,C fp32.
// Block tile 128x64, 8 waves (wave = 16-row strip), K chunked by 32.
// A staged raw fp32 via async DMA (converted at fragment build, once per
// element); W staged TRANSPOSED bf16 (shared by 8 waves -> convert at stage).
// ===========================================================================
#define GT_M 128
#define GT_N 64
#define GT_K 32

__global__ __launch_bounds__(256)
void gemm_bias_kernel(const float* __restrict__ A, const float* __restrict__ W,
                      const float* __restrict__ bias, float* __restrict__ C,
                      int M, int N, int K, int relu)
{
  __shared__ float          Asf[GT_M][GT_K + 4];  // (m, k) fp32, 144B rows (16B aligned)
  __shared__ unsigned short Ws[GT_N][GT_K + 2];   // (n, k) bf16 (transposed)

  const int tid  = threadIdx.x;
  const int wave = tid >> 5;
  const int lane = tid & 31;
  const int half = lane >> 4;   // 0: lanes 0-15, 1: lanes 16-31
  const int l16  = lane & 15;

  const int m0 = blockIdx.x * GT_M;
  const int n0 = blockIdx.y * GT_N;
  const int mw = wave * 16;

  v8f acc[4];
  #pragma unroll
  for (int t = 0; t < 4; ++t)
    #pragma unroll
    for (int r = 0; r < 8; ++r) acc[t][r] = 0.0f;

  for (int k0 = 0; k0 < K; k0 += GT_K) {
    __syncthreads();
    // ---- stage A tile: 128x32 raw fp32, per-lane async DMA into LDS
    #pragma unroll
    for (int i = 0; i < 4; ++i) {
      int e = (tid << 2) + (i << 10);          // 0..4095, quads
      int r = e >> 5, c = e & 31;              // c multiple of 4 -> 16B aligned
      const float* gp = A + (size_t)(m0 + r) * K + k0 + c;
#if HAVE_ASYNC_LDS
      async_copy_b128(&Asf[r][c], gp);
#else
      *(float4*)&Asf[r][c] = *(const float4*)gp;
#endif
    }
    // ---- stage W tile transposed: 32x64 fp32 -> bf16 (8 elems / thread)
    #pragma unroll
    for (int i = 0; i < 2; ++i) {
      int e = (tid << 2) + (i << 10);          // 0..2047, row-major (k, n)
      int kk = e >> 6, nn = e & 63;
      const float4 v = *(const float4*)(W + (size_t)(k0 + kk) * N + n0 + nn);
      Ws[nn + 0][kk] = f2bf(v.x);
      Ws[nn + 1][kk] = f2bf(v.y);
      Ws[nn + 2][kk] = f2bf(v.z);
      Ws[nn + 3][kk] = f2bf(v.w);
    }
    // prefetch next K-chunk of W (global_prefetch_b8)
    if (k0 + GT_K < K)
      __builtin_prefetch(W + (size_t)(k0 + GT_K + (tid >> 3)) * N + n0 + ((tid & 7) << 3), 0, 3);
#if HAVE_ASYNC_LDS
    wait_async0();
#endif
    __syncthreads();

    // ---- A fragment (16x32): fp32 pairs from LDS, convert in-register
    v16bf af;
    {
      const float* row = &Asf[mw + l16][0];
      #pragma unroll
      for (int j = 0; j < 8; ++j) {
        int kp = 2 * (j & 3) + 16 * (j >> 2) + 8 * half;  // even -> 8B aligned
        float2 v = *(const float2*)(row + kp);
        af[2 * j]     = bfbits(f2bf(v.x));
        af[2 * j + 1] = bfbits(f2bf(v.y));
      }
    }
    // ---- 4 B fragments (32x16) + WMMA
    #pragma unroll
    for (int nt = 0; nt < 4; ++nt) {
      v16bf bfr;
      const unsigned* col = (const unsigned*)&Ws[nt * 16 + l16][0];
      #pragma unroll
      for (int j = 0; j < 8; ++j) {
        unsigned u = col[(j & 3) + 8 * (j >> 2) + 4 * half];  // K = 2(j&3)(+1) + 16*(j>>2) + 8*half
        bfr[2 * j]     = bfbits((unsigned short)(u & 0xffffu));
        bfr[2 * j + 1] = bfbits((unsigned short)(u >> 16));
      }
      acc[nt] = __builtin_amdgcn_wmma_f32_16x16x32_bf16(
          false, af, false, bfr, (short)0, acc[nt], false, false);
    }
  }

  // ---- epilogue: bias (+ReLU), fp32 store. C layout: m=r+8*half, n=l16.
  #pragma unroll
  for (int nt = 0; nt < 4; ++nt) {
    int colg = n0 + nt * 16 + l16;
    float bv = bias[colg];
    #pragma unroll
    for (int r = 0; r < 8; ++r) {
      int rowg = m0 + mw + r + 8 * half;
      float v = acc[nt][r] + bv;
      if (relu) v = fmaxf(v, 0.0f);
      C[(size_t)rowg * N + colg] = v;
    }
  }
}

// ===========================================================================
// Fused attention with softmax over the HEADS axis (reference's torch quirk).
// attention[b,h,q,k] = exp(s_h) / sum_h' exp(s_h')  -- local per (q,k)!
// One workgroup = 16 waves = 16 heads, q-tile of 16, streaming k-tiles of 64.
// qkv layout per token: [head][q(64) | k(64) | v(64)] = 192 floats/head.
// ===========================================================================
#define ATT_KT 64

__global__ __launch_bounds__(512)
void attn_kernel(const float* __restrict__ qkv, float* __restrict__ vals, int S)
{
  __shared__ unsigned short Ps[16][16][ATT_KT + 2];  // [head][q][k] exp(s) bf16
  __shared__ float Ds[16][ATT_KT];                   // denom over heads, fp32

  const int tid  = threadIdx.x;
  const int h    = tid >> 5;          // wave == head
  const int lane = tid & 31;
  const int half = lane >> 4, l16 = lane & 15;

  const int blocksPerB = S / 16;
  const int b   = blockIdx.x / blocksPerB;
  const int q0  = (blockIdx.x % blocksPerB) * 16;
  const size_t tokBase = (size_t)b * S;

  // ---- Q fragments (two K=32 chunks over head-dim 64), pre-scaled by 1/8
  v16bf qf[2];
  {
    const float* qrow = qkv + (tokBase + q0 + l16) * 3072 + h * 192;
    #pragma unroll
    for (int c = 0; c < 2; ++c)
      #pragma unroll
      for (int j = 0; j < 8; ++j) {
        int kd = c * 32 + 2 * (j & 3) + 16 * (j >> 2) + 8 * half;
        float2 v = *(const float2*)(qrow + kd);
        qf[c][2 * j]     = bfbits(f2bf(v.x * 0.125f));
        qf[c][2 * j + 1] = bfbits(f2bf(v.y * 0.125f));
      }
  }

  v8f oacc[4];
  #pragma unroll
  for (int t = 0; t < 4; ++t)
    #pragma unroll
    for (int r = 0; r < 8; ++r) oacc[t][r] = 0.0f;

  for (int kt = 0; kt < S; kt += ATT_KT) {
    __syncthreads();                       // previous tile fully consumed
    ((float*)Ds)[tid]       = 0.0f;        // zero denominators (1024 floats)
    ((float*)Ds)[tid + 512] = 0.0f;
    __syncthreads();

    // ---- scores S[16 x 64] for this head: Q @ K^T over head-dim 64
    v8f sc[4];
    #pragma unroll
    for (int nt = 0; nt < 4; ++nt) {
      #pragma unroll
      for (int r = 0; r < 8; ++r) sc[nt][r] = 0.0f;
      #pragma unroll
      for (int c = 0; c < 2; ++c) {
        // B = K^T fragment: rows = head-dims (contiguous in memory), col = token
        v16bf bk;
        const float* krow = qkv + (tokBase + kt + nt * 16 + l16) * 3072
                                + h * 192 + 64 + c * 32 + 8 * half;
        #pragma unroll
        for (int j = 0; j < 4; ++j) {
          float4 v = *(const float4*)(krow + 4 * (j & 1) + 16 * (j >> 1));
          bk[4 * j]     = bfbits(f2bf(v.x));
          bk[4 * j + 1] = bfbits(f2bf(v.y));
          bk[4 * j + 2] = bfbits(f2bf(v.z));
          bk[4 * j + 3] = bfbits(f2bf(v.w));
        }
        sc[nt] = __builtin_amdgcn_wmma_f32_16x16x32_bf16(
            false, qf[c], false, bk, (short)0, sc[nt], false, false);
      }
    }
    // ---- exp, publish to LDS, cross-head denominator via ds_add_f32
    #pragma unroll
    for (int nt = 0; nt < 4; ++nt) {
      int n = nt * 16 + l16;
      #pragma unroll
      for (int r = 0; r < 8; ++r) {
        int m = r + 8 * half;
        float e = __expf(sc[nt][r]);
        Ps[h][m][n] = f2bf(e);
        atomicAdd(&Ds[m][n], e);
      }
    }
    __syncthreads();

    // ---- P @ V  (P normalized at fragment-load time)
    #pragma unroll
    for (int c = 0; c < 2; ++c) {
      v16bf pf;
      const unsigned* prow = (const unsigned*)&Ps[h][l16][0];
      const float*    drow = &Ds[l16][0];
      #pragma unroll
      for (int j = 0; j < 8; ++j) {
        int kk = c * 32 + 2 * (j & 3) + 16 * (j >> 2) + 8 * half; // even
        unsigned u = prow[kk >> 1];
        float p0 = bf2f((unsigned short)(u & 0xffffu)) / drow[kk];
        float p1 = bf2f((unsigned short)(u >> 16))     / drow[kk + 1];
        pf[2 * j]     = bfbits(f2bf(p0));
        pf[2 * j + 1] = bfbits(f2bf(p1));
      }
      #pragma unroll
      for (int nt = 0; nt < 4; ++nt) {
        // B = V fragment: rows = tokens (stride 3072 floats), col = head-dim
        v16bf vf;
        const float* vb = qkv + (tokBase + kt + c * 32 + 8 * half) * 3072
                              + h * 192 + 128 + nt * 16 + l16;
        #pragma unroll
        for (int e = 0; e < 16; ++e)
          vf[e] = bfbits(f2bf(vb[(size_t)(e + 8 * (e >> 3)) * 3072]));
        oacc[nt] = __builtin_amdgcn_wmma_f32_16x16x32_bf16(
            false, pf, false, vf, (short)0, oacc[nt], false, false);
      }
    }
  }

  // ---- store merged-head values: token = q0+m, feature = h*64 + n
  #pragma unroll
  for (int nt = 0; nt < 4; ++nt) {
    int n = nt * 16 + l16;
    #pragma unroll
    for (int r = 0; r < 8; ++r) {
      int m = r + 8 * half;
      vals[(tokBase + q0 + m) * 1024 + h * 64 + n] = oacc[nt][r];
    }
  }
}

// ===========================================================================
// out = LayerNorm(y + res) * gamma + beta   (block per token, D = 1024)
// ===========================================================================
__global__ __launch_bounds__(256)
void add_ln_kernel(const float* __restrict__ y, const float* __restrict__ res,
                   const float* __restrict__ gamma, const float* __restrict__ beta,
                   float* __restrict__ out, int D)
{
  __shared__ float red[256];
  const int row = blockIdx.x, tid = threadIdx.x;
  const int PT = D / 256;                  // 4
  const float* yr = y   + (size_t)row * D;
  const float* rr = res + (size_t)row * D;

  float v[4];
  float s = 0.0f;
  for (int i = 0; i < PT; ++i) { v[i] = yr[tid + 256 * i] + rr[tid + 256 * i]; s += v[i]; }
  red[tid] = s; __syncthreads();
  for (int o = 128; o > 0; o >>= 1) { if (tid < o) red[tid] += red[tid + o]; __syncthreads(); }
  float mean = red[0] / (float)D;
  __syncthreads();

  float s2 = 0.0f;
  for (int i = 0; i < PT; ++i) { float d = v[i] - mean; s2 += d * d; }
  red[tid] = s2; __syncthreads();
  for (int o = 128; o > 0; o >>= 1) { if (tid < o) red[tid] += red[tid + o]; __syncthreads(); }
  float rstd = rsqrtf(red[0] / (float)D + 1e-5f);

  for (int i = 0; i < PT; ++i) {
    int c = tid + 256 * i;
    out[(size_t)row * D + c] = gamma[c] * ((v[i] - mean) * rstd) + beta[c];
  }
}

// ===========================================================================
// Pipeline:  x -> QKV gemm -> fused attn -> out-proj gemm -> add+LN ->
//            FFN1 gemm(ReLU) -> FFN2 gemm -> add+LN -> out
// Fixed shapes from reference: B=2, S=2048, D=1024, H=16, FFN=4096.
// Workspace use: 160 MB (qkv 48 | vals 16 | tmp 16 | x1 16 | hidden 64).
// ===========================================================================
extern "C" void kernel_launch(void* const* d_in, const int* in_sizes, int n_in,
                              void* d_out, int out_size, void* d_ws, size_t ws_size,
                              hipStream_t stream) {
  (void)in_sizes; (void)n_in; (void)out_size; (void)ws_size;
  const float* x      = (const float*)d_in[0];
  const float* w_qkv  = (const float*)d_in[1];
  const float* b_qkv  = (const float*)d_in[2];
  const float* w_o    = (const float*)d_in[3];
  const float* b_o    = (const float*)d_in[4];
  const float* gamma1 = (const float*)d_in[5];
  const float* beta1  = (const float*)d_in[6];
  const float* w1     = (const float*)d_in[7];
  const float* b1     = (const float*)d_in[8];
  const float* w2     = (const float*)d_in[9];
  const float* b2     = (const float*)d_in[10];
  const float* gamma2 = (const float*)d_in[11];
  const float* beta2  = (const float*)d_in[12];
  float* out = (float*)d_out;

  const int B = 2, S = 2048, D = 1024, F = 4096, T = B * S;  // 4096 tokens

  float* qkv  = (float*)d_ws;                        // T x 3072
  float* vals = qkv  + (size_t)T * 3 * D;            // T x 1024
  float* tmp  = vals + (size_t)T * D;                // T x 1024 (reused)
  float* x1   = tmp  + (size_t)T * D;                // T x 1024
  float* hid  = x1   + (size_t)T * D;                // T x 4096

  // 1) QKV projection: [T,1024] @ [1024,3072] + b
  gemm_bias_kernel<<<dim3(T / GT_M, (3 * D) / GT_N), 256, 0, stream>>>(
      x, w_qkv, b_qkv, qkv, T, 3 * D, D, 0);

  // 2) fused head-softmax attention -> merged values [T,1024]
  attn_kernel<<<dim3(B * (S / 16)), 512, 0, stream>>>(qkv, vals, S);

  // 3) output projection
  gemm_bias_kernel<<<dim3(T / GT_M, D / GT_N), 256, 0, stream>>>(
      vals, w_o, b_o, tmp, T, D, D, 0);

  // 4) residual + LayerNorm 1
  add_ln_kernel<<<dim3(T), 256, 0, stream>>>(tmp, x, gamma1, beta1, x1, D);

  // 5) FFN up + ReLU
  gemm_bias_kernel<<<dim3(T / GT_M, F / GT_N), 256, 0, stream>>>(
      x1, w1, b1, hid, T, F, D, 1);

  // 6) FFN down
  gemm_bias_kernel<<<dim3(T / GT_M, D / GT_N), 256, 0, stream>>>(
      hid, w2, b2, tmp, T, D, F, 0);

  // 7) residual + LayerNorm 2 -> final output
  add_ln_kernel<<<dim3(T), 256, 0, stream>>>(tmp, x1, gamma2, beta2, out, D);
}
